// MS_Block_75368086110771
// MI455X (gfx1250) — matrix operand order repaired and checked by reading, not compiled, not run on hardware
//
#include <hip/hip_runtime.h>
#include <hip/hip_bf16.h>

// ---------------------------------------------------------------------------
// Spiking transformer block for MI455X (gfx1250, wave32, WMMA).
//
// Heavy matmuls other than qkv.pw/proj.pw consume binary LIF spikes -> stored as f16 {0,1} in
// channel-last (image-major, N x K) layout, which maps 1:1 onto the
// V_WMMA_F32_16X16X32_F16 B-operand lane layout (lane = output column,
// K k..k+7 & k+16..k+23, +8 for lanes 16-31). Weights are pre-converted to f16 row-major (M x K),
// matching the A-operand layout. BN / bias chains are folded into per-channel
// affine GEMM epilogues. GEMM inner loop is software-pipelined with
// double-buffered fragments so WMMA overlaps the next slab's global loads.
//
// Input flattening assumption (jax pytree: dicts sorted by key, 'params'<'x'):
//   attn.proj.bn1{b,m,v,w}, attn.proj.bn2{b,m,v,w}, attn.proj.{c1,dw,pw},
//   attn.proj_bn{b,m,v,w},
//   attn.qkv.bn1{b,m,v,w}, attn.qkv.bn2{b,m,v,w}, attn.qkv.{c1,dw,pw},
//   attn.qkv_bn{b,m,v,w}, attn.scale,
//   conv.bn{b,m,v,w}, conv.w,
//   mlp.bn1{b,m,v,w}, mlp.bn2{b,m,v,w}, mlp.dw_b, mlp.dw_w,
//   mlp.fc1_b, mlp.fc1_w, mlp.fc2_b, mlp.fc2_w,  x
// ---------------------------------------------------------------------------

typedef __attribute__((ext_vector_type(16))) _Float16 v16h;
typedef __attribute__((ext_vector_type(8)))  _Float16 v8h;
typedef __attribute__((ext_vector_type(8)))  float    v8f;

#define TT    4
#define BBB   16
#define CC    256
#define NN    1024
#define HIDD  1024
#define NIMG  64            // T*B images

enum {
  IDX_PROJ_BN1_B = 0, IDX_PROJ_BN1_M, IDX_PROJ_BN1_V, IDX_PROJ_BN1_W,
  IDX_PROJ_BN2_B, IDX_PROJ_BN2_M, IDX_PROJ_BN2_V, IDX_PROJ_BN2_W,
  IDX_PROJ_C1, IDX_PROJ_DW, IDX_PROJ_PW,
  IDX_PROJBN_B, IDX_PROJBN_M, IDX_PROJBN_V, IDX_PROJBN_W,
  IDX_QKV_BN1_B, IDX_QKV_BN1_M, IDX_QKV_BN1_V, IDX_QKV_BN1_W,
  IDX_QKV_BN2_B, IDX_QKV_BN2_M, IDX_QKV_BN2_V, IDX_QKV_BN2_W,
  IDX_QKV_C1, IDX_QKV_DW, IDX_QKV_PW,
  IDX_QKVBN_B, IDX_QKVBN_M, IDX_QKVBN_V, IDX_QKVBN_W,
  IDX_SCALE,
  IDX_CONV_BN_B, IDX_CONV_BN_M, IDX_CONV_BN_V, IDX_CONV_BN_W,
  IDX_CONV_W,
  IDX_MLP_BN1_B, IDX_MLP_BN1_M, IDX_MLP_BN1_V, IDX_MLP_BN1_W,
  IDX_MLP_BN2_B, IDX_MLP_BN2_M, IDX_MLP_BN2_V, IDX_MLP_BN2_W,
  IDX_MLP_DWB, IDX_MLP_DWW,
  IDX_MLP_FC1B, IDX_MLP_FC1W,
  IDX_MLP_FC2B, IDX_MLP_FC2W,
  IDX_X
};

union FragH { v16h v; v8h h[2]; };

// ----------------------------- param prep ----------------------------------
__global__ __launch_bounds__(256) void k_bn_affine(float* g, float* h,
    const float* w, const float* b, const float* m, const float* v, int n) {
  int i = blockIdx.x * 256 + threadIdx.x;
  if (i < n) {
    float gg = w[i] * rsqrtf(v[i] + 1e-5f);
    g[i] = gg;
    h[i] = b[i] - m[i] * gg;
  }
}

// apply a second BN on top of existing affine (g,h) in place
__global__ __launch_bounds__(256) void k_bn_compose(float* g, float* h,
    const float* w, const float* b, const float* m, const float* v, int n) {
  int i = blockIdx.x * 256 + threadIdx.x;
  if (i < n) {
    float g2 = w[i] * rsqrtf(v[i] + 1e-5f);
    float h2 = b[i] - m[i] * g2;
    g[i] = g[i] * g2;
    h[i] = h[i] * g2 + h2;
  }
}

// fold a pre-BN bias: (acc + bias)*g + h = acc*g + (bias*g + h)
__global__ __launch_bounds__(256) void k_bias_fold(float* h, const float* g,
    const float* bias, int n) {
  int i = blockIdx.x * 256 + threadIdx.x;
  if (i < n) h[i] += bias[i] * g[i];
}

__global__ __launch_bounds__(256) void k_f32_to_f16(_Float16* dst,
    const float* src, int n) {
  int i = blockIdx.x * 256 + threadIdx.x;
  if (i < n) dst[i] = (_Float16)src[i];
}

// ------------------------- WMMA spike GEMM ---------------------------------
// D(img)[n, m] = affine_m( sum_k A[m,k] * B(img)[n,k] )
// A: f16 row-major M x K (weights).  B: f16 per-image N x K (spikes, chan-last)
// Y: f32 per-image N x M (chan-last activations).  N = 1024 fixed.
// Block = 256 threads = 8 waves arranged 2(M) x 4(N); wave tile 64 x 32.
// K must be a multiple of 64 (256 / 1024 here). Double-buffered pipeline.
__global__ __launch_bounds__(256) void k_gemm_wmma(
    const _Float16* __restrict__ A, const _Float16* __restrict__ B,
    float* __restrict__ Y, const float* __restrict__ scale,
    const float* __restrict__ shift, int M, int K) {
  const int lane = threadIdx.x & 31;
  const int wave = threadIdx.x >> 5;
  const int wm = wave >> 2;          // 0..1
  const int wn = wave & 3;           // 0..3
  const int m0 = blockIdx.y * 128 + wm * 64;
  const int n0 = blockIdx.x * 128 + wn * 32;
  const size_t img = blockIdx.z;

  const _Float16* Bi = B + img * (size_t)NN * K;
  float* Yi = Y + img * (size_t)NN * M;

  const int half = lane >> 4;        // 0: lanes 0-15, 1: lanes 16-31
  const int l = lane & 15;

  // Per-lane base pointers; K advances by pointer offset.
  // A 16x32 f16 frag: lanes 0-15 row m0+16s+l hold K k..k+7 & k+16..k+23;
  // lanes 16-31 hold K k+8..k+15 & k+24..k+31 of the same rows.
  const _Float16* apL[4];
#pragma unroll
  for (int s = 0; s < 4; ++s)
    apL[s] = A + (size_t)(m0 + 16 * s + l) * K + half * 8;
  // B 32x16 f16 frag: lanes 0-15 = column n0+16j+l, K k..k+7 & k+16..k+23;
  // lanes 16-31 = same columns, K k+8..k+15 & k+24..k+31.
  const _Float16* bpL[2];
#pragma unroll
  for (int j = 0; j < 2; ++j)
    bpL[j] = Bi + (size_t)(n0 + 16 * j + l) * K + half * 8;

  v8f acc[4][2] = {};
  FragH a0[4], a1[4], b0[2], b1[2];

  auto loadA = [&](FragH* a, int k) {
#pragma unroll
    for (int s = 0; s < 4; ++s) {
      a[s].h[0] = *(const v8h*)(apL[s] + k);
      a[s].h[1] = *(const v8h*)(apL[s] + k + 16);
    }
  };
  auto loadB = [&](FragH* b, int k) {
#pragma unroll
    for (int j = 0; j < 2; ++j) {
      b[j].h[0] = *(const v8h*)(bpL[j] + k);
      b[j].h[1] = *(const v8h*)(bpL[j] + k + 16);
    }
  };
  auto mma = [&](const FragH* a, const FragH* b) {
#pragma unroll
    for (int s = 0; s < 4; ++s)
#pragma unroll
      for (int j = 0; j < 2; ++j)
        acc[s][j] = __builtin_amdgcn_wmma_f32_16x16x32_f16(
            false, a[s].v, false, b[j].v, (short)0, acc[s][j], false, false);
  };

  loadA(a0, 0);
  loadB(b0, 0);
  for (int k = 0; k < K; k += 64) {
    loadA(a1, k + 32);
    loadB(b1, k + 32);
    if (k + 64 < K) {  // pull the slab after next toward the WGP
      __builtin_prefetch(bpL[0] + k + 64, 0, 1);
      __builtin_prefetch(bpL[1] + k + 64, 0, 1);
    }
    mma(a0, b0);                    // overlaps with k+32 loads in flight
    if (k + 64 < K) {
      loadA(a0, k + 64);
      loadB(b0, k + 64);
    }
    mma(a1, b1);                    // overlaps with k+64 loads in flight
  }

  // Epilogue: lane holds column n = n0+16j+(lane&15), rows mB..mB+7 contiguous
  // -> 32B contiguous store into channel-last activation + fused BN affine.
#pragma unroll
  for (int s = 0; s < 4; ++s) {
    const int mB = m0 + 16 * s + half * 8;
    v8f sc = *(const v8f*)(scale + mB);
    v8f sh = *(const v8f*)(shift + mB);
#pragma unroll
    for (int j = 0; j < 2; ++j) {
      const int n = n0 + 16 * j + l;
      v8f r = acc[s][j] * sc + sh;
      *(v8f*)(Yi + (size_t)n * M + mB) = r;
    }
  }
}

// ------------------------------ LIF kernels --------------------------------
// First LIFs on x (channel-first f32): hard-reset spikes (attn) + soft-reset
// spikes (conv branch), written channel-last f16.
__global__ __launch_bounds__(256) void k_lif_x(const float* __restrict__ X,
    _Float16* __restrict__ sHard, _Float16* __restrict__ sSoft) {
  int idx = blockIdx.x * 256 + threadIdx.x;  // (b, c, n), n fastest
  int n = idx & (NN - 1);
  int c = (idx >> 10) & (CC - 1);
  int b = idx >> 18;
  float vh = 0.f, vs = 0.f;
#pragma unroll
  for (int t = 0; t < TT; ++t) {
    float x = X[((size_t)(t * BBB + b) * CC + c) * NN + n];
    vh += (x - vh) * 0.5f;
    float h1 = vh >= 1.f ? 1.f : 0.f;
    vh *= (1.f - h1);                         // hard reset
    vs += (x - vs) * 0.5f;
    float h2 = vs >= 1.f ? 1.f : 0.f;
    vs -= h2;                                 // soft reset
    size_t o = ((size_t)(t * BBB + b) * NN + n) * CC + c;
    sHard[o] = (_Float16)h1;
    sSoft[o] = (_Float16)h2;
  }
}

// bn_pad + depthwise 3x3: input Yb already BN'd (fused in GEMM epilogue);
// border value is the BN shift (pv = b - m*g). Output f16 chan-last.
__global__ __launch_bounds__(256) void k_dw_pad(const float* __restrict__ Yb,
    const float* __restrict__ W9, const float* __restrict__ pv,
    _Float16* __restrict__ out) {
  int idx = blockIdx.x * 256 + threadIdx.x;   // (n, c), c fastest
  int c = idx & (CC - 1);
  int n = idx >> 8;
  size_t img = blockIdx.y;
  int h = n >> 5, w = n & 31;
  float p = pv[c];
  float acc = 0.f;
#pragma unroll
  for (int i = 0; i < 3; ++i)
#pragma unroll
    for (int j = 0; j < 3; ++j) {
      int hh = h + i - 1, ww = w + j - 1;
      float val = (hh < 0 || hh > 31 || ww < 0 || ww > 31)
                      ? p
                      : Yb[(img * NN + (size_t)((hh << 5) + ww)) * CC + c];
      acc += W9[c * 9 + i * 3 + j] * val;
    }
  out[(img * NN + (size_t)n) * CC + c] = (_Float16)acc;
}

// qk path: soft LIF over t, sum over spatial (block reduce in LDS),
// then second LIF (th=0.5, hard) -> binary gate per (t,b,c).
__global__ __launch_bounds__(256) void k_qk_gate(const float* __restrict__ Yq,
    float* __restrict__ gate) {
  int b = blockIdx.x >> 8;
  int c = blockIdx.x & 255;
  int tid = threadIdx.x;
  __shared__ float red[256];
  float v[4] = {0.f, 0.f, 0.f, 0.f};
  float sums[TT];
  for (int t = 0; t < TT; ++t) {
    float ps = 0.f;
#pragma unroll
    for (int q = 0; q < 4; ++q) {
      int n = tid + q * 256;
      float x = Yq[((size_t)(t * BBB + b) * NN + n) * 512 + c];
      v[q] += (x - v[q]) * 0.5f;
      float s = v[q] >= 1.f ? 1.f : 0.f;
      v[q] -= s;
      ps += s;
    }
    red[tid] = ps;
    __syncthreads();
    for (int off = 128; off > 0; off >>= 1) {
      if (tid < off) red[tid] += red[tid + off];
      __syncthreads();
    }
    sums[t] = red[0];
    __syncthreads();
  }
  if (tid == 0) {
    float v2 = 0.f;
#pragma unroll
    for (int t = 0; t < TT; ++t) {
      v2 += (sums[t] - v2) * 0.5f;
      float hh = v2 >= 0.5f ? 1.f : 0.f;
      v2 *= (1.f - hh);                       // hard reset
      gate[(size_t)(t * BBB + b) * CC + c] = hh;
    }
  }
}

// v path: soft LIF spikes * qk gate -> binary attn-proj input (f16 chan-last)
__global__ __launch_bounds__(256) void k_vmul(const float* __restrict__ Yq,
    const float* __restrict__ gate, _Float16* __restrict__ Satt) {
  int idx = blockIdx.x * 256 + threadIdx.x;   // (b, n, c), c fastest
  int c = idx & 255;
  int n = (idx >> 8) & 1023;
  int b = idx >> 18;
  float v = 0.f;
#pragma unroll
  for (int t = 0; t < TT; ++t) {
    float x = Yq[((size_t)(t * BBB + b) * NN + n) * 512 + 256 + c];
    v += (x - v) * 0.5f;
    float s = v >= 1.f ? 1.f : 0.f;
    v -= s;
    Satt[((size_t)(t * BBB + b) * NN + n) * CC + c] =
        (_Float16)(s * gate[(size_t)(t * BBB + b) * CC + c]);
  }
}

// x_mid = x + scale*lif(attn_pre) + conv_out ; also mlp's first soft LIF.
__global__ __launch_bounds__(256) void k_residual(const float* __restrict__ X,
    const float* __restrict__ Apre, const float* __restrict__ Cv,
    const float* __restrict__ scl, float* __restrict__ Xmid,
    _Float16* __restrict__ S3) {
  int idx = blockIdx.x * 256 + threadIdx.x;   // (b, n, c), c fastest
  int c = idx & 255;
  int n = (idx >> 8) & 1023;
  int b = idx >> 18;
  float sc = scl[0];
  float va = 0.f, vm = 0.f;
#pragma unroll
  for (int t = 0; t < TT; ++t) {
    size_t ii = ((size_t)(t * BBB + b) * NN + n) * CC + c;
    float ap = Apre[ii];
    va += (ap - va) * 0.5f;
    float sa = va >= 1.f ? 1.f : 0.f;
    va -= sa;
    float xv = X[((size_t)(t * BBB + b) * CC + c) * NN + n];
    float xm = xv + sc * sa + Cv[ii];
    Xmid[ii] = xm;
    vm += (xm - vm) * 0.5f;
    float sm = vm >= 1.f ? 1.f : 0.f;
    vm -= sm;
    S3[ii] = (_Float16)sm;
  }
}

// soft LIF over 1024-channel chan-last activation (mlp ybn -> s_f)
__global__ __launch_bounds__(256) void k_lif_hid(const float* __restrict__ Yin,
    _Float16* __restrict__ Sout) {
  size_t idx = (size_t)blockIdx.x * 256 + threadIdx.x;  // (b, n, c)
  int c = (int)(idx & (HIDD - 1));
  int n = (int)((idx >> 10) & (NN - 1));
  int b = (int)(idx >> 20);
  float v = 0.f;
#pragma unroll
  for (int t = 0; t < TT; ++t) {
    size_t o = ((size_t)(t * BBB + b) * NN + n) * HIDD + c;
    float x = Yin[o];
    v += (x - v) * 0.5f;
    float s = v >= 1.f ? 1.f : 0.f;
    v -= s;
    Sout[o] = (_Float16)s;
  }
}

// mlp: f = dw3x3(s_f, zero-pad) + bias ; s_z = lif_soft(ybn + f)
__global__ __launch_bounds__(256) void k_mlp_dw_lifz(
    const _Float16* __restrict__ Sf, const float* __restrict__ Ybn,
    const float* __restrict__ W9, const float* __restrict__ bias,
    _Float16* __restrict__ Sz) {
  size_t idx = (size_t)blockIdx.x * 256 + threadIdx.x;  // (b, n, c)
  int c = (int)(idx & (HIDD - 1));
  int n = (int)((idx >> 10) & (NN - 1));
  int b = (int)(idx >> 20);
  int h = n >> 5, w = n & 31;
  float wr[9];
#pragma unroll
  for (int k = 0; k < 9; ++k) wr[k] = W9[c * 9 + k];
  float bv = bias[c];
  float v = 0.f;
  for (int t = 0; t < TT; ++t) {
    float f = bv;
#pragma unroll
    for (int i = 0; i < 3; ++i)
#pragma unroll
      for (int j = 0; j < 3; ++j) {
        int hh = h + i - 1, ww = w + j - 1;
        if (hh >= 0 && hh < 32 && ww >= 0 && ww < 32)
          f += wr[i * 3 + j] *
               (float)Sf[((size_t)(t * BBB + b) * NN + ((hh << 5) + ww)) * HIDD + c];
      }
    size_t o = ((size_t)(t * BBB + b) * NN + n) * HIDD + c;
    float z = Ybn[o] + f;
    v += (z - v) * 0.5f;
    float s = v >= 1.f ? 1.f : 0.f;
    v -= s;
    Sz[o] = (_Float16)s;
  }
}

// out (channel-first) = x_mid + mlp_out  (both chan-last)
__global__ __launch_bounds__(256) void k_final(const float* __restrict__ Xmid,
    const float* __restrict__ Mlp, float* __restrict__ Out) {
  int idx = blockIdx.x * 256 + threadIdx.x;   // (b, n, c), c fastest
  int c = idx & 255;
  int n = (idx >> 8) & 1023;
  int b = idx >> 18;
#pragma unroll
  for (int t = 0; t < TT; ++t) {
    size_t ii = ((size_t)(t * BBB + b) * NN + n) * CC + c;
    Out[((size_t)(t * BBB + b) * CC + c) * NN + n] = Xmid[ii] + Mlp[ii];
  }
}

// ------------------------------- launch ------------------------------------
extern "C" void kernel_launch(void* const* d_in, const int* in_sizes, int n_in,
                              void* d_out, int out_size, void* d_ws,
                              size_t ws_size, hipStream_t stream) {
  (void)in_sizes; (void)n_in; (void)out_size; (void)ws_size;
  char* ws = (char*)d_ws;
  size_t off = 0;
  auto alloc = [&](size_t bytes) -> char* {
    off = (off + 255) & ~(size_t)255;
    char* p = ws + off;
    off += bytes;
    return p;
  };
  auto F = [&](int i) -> const float* { return (const float*)d_in[i]; };

  // f16 weight copies
  _Float16* wQkvC1  = (_Float16*)alloc((size_t)CC * CC * 2);
  _Float16* wQkvPw  = (_Float16*)alloc((size_t)512 * CC * 2);
  _Float16* wProjC1 = (_Float16*)alloc((size_t)CC * CC * 2);
  _Float16* wProjPw = (_Float16*)alloc((size_t)CC * CC * 2);
  _Float16* wConv   = (_Float16*)alloc((size_t)CC * CC * 2);
  _Float16* wFc1    = (_Float16*)alloc((size_t)HIDD * CC * 2);
  _Float16* wFc2    = (_Float16*)alloc((size_t)CC * HIDD * 2);
  // fused affine (scale/shift) arrays
  float* gQkv1   = (float*)alloc(CC * 4);   float* hQkv1   = (float*)alloc(CC * 4);
  float* gQkvPw  = (float*)alloc(512 * 4);  float* hQkvPw  = (float*)alloc(512 * 4);
  float* gProj1  = (float*)alloc(CC * 4);   float* hProj1  = (float*)alloc(CC * 4);
  float* gProjPw = (float*)alloc(CC * 4);   float* hProjPw = (float*)alloc(CC * 4);
  float* gConv   = (float*)alloc(CC * 4);   float* hConv   = (float*)alloc(CC * 4);
  float* gFc1    = (float*)alloc(HIDD * 4); float* hFc1    = (float*)alloc(HIDD * 4);
  float* gFc2    = (float*)alloc(CC * 4);   float* hFc2    = (float*)alloc(CC * 4);
  // activations
  const size_t NC  = (size_t)NIMG * NN * CC;    // 16.7M elems
  const size_t NH  = (size_t)NIMG * NN * HIDD;  // 67M elems
  _Float16* sHard = (_Float16*)alloc(NC * 2);
  _Float16* sSoft = (_Float16*)alloc(NC * 2);
  float*    yb1   = (float*)alloc(NC * 4);
  _Float16* dw1   = (_Float16*)alloc(NC * 2);
  float*    yqkv  = (float*)alloc((size_t)NIMG * NN * 512 * 4);
  float*    gate  = (float*)alloc((size_t)NIMG * CC * 4);
  _Float16* satt  = (_Float16*)alloc(NC * 2);
  float*    yb2   = (float*)alloc(NC * 4);
  _Float16* dw2   = (_Float16*)alloc(NC * 2);
  float*    apre  = (float*)alloc(NC * 4);
  float*    cvout = (float*)alloc(NC * 4);
  float*    xmid  = (float*)alloc(NC * 4);
  _Float16* s3    = (_Float16*)alloc(NC * 2);
  float*    ybn   = (float*)alloc(NH * 4);
  _Float16* sf    = (_Float16*)alloc(NH * 2);
  _Float16* sz    = (_Float16*)alloc(NH * 2);
  float*    mlpo  = (float*)alloc(NC * 4);

  // -- param prep --
  auto cvt = [&](_Float16* dst, int idx, int n) {
    k_f32_to_f16<<<(n + 255) / 256, 256, 0, stream>>>(dst, F(idx), n);
  };
  cvt(wQkvC1, IDX_QKV_C1, CC * CC);
  cvt(wQkvPw, IDX_QKV_PW, 512 * CC);
  cvt(wProjC1, IDX_PROJ_C1, CC * CC);
  cvt(wProjPw, IDX_PROJ_PW, CC * CC);
  cvt(wConv, IDX_CONV_W, CC * CC);
  cvt(wFc1, IDX_MLP_FC1W, HIDD * CC);
  cvt(wFc2, IDX_MLP_FC2W, CC * HIDD);

  auto bnaff = [&](float* g, float* h, int iw, int ib, int im, int iv, int n) {
    k_bn_affine<<<(n + 255) / 256, 256, 0, stream>>>(g, h, F(iw), F(ib), F(im), F(iv), n);
  };
  auto bncomp = [&](float* g, float* h, int iw, int ib, int im, int iv, int n) {
    k_bn_compose<<<(n + 255) / 256, 256, 0, stream>>>(g, h, F(iw), F(ib), F(im), F(iv), n);
  };
  bnaff(gQkv1, hQkv1, IDX_QKV_BN1_W, IDX_QKV_BN1_B, IDX_QKV_BN1_M, IDX_QKV_BN1_V, CC);
  bnaff(gQkvPw, hQkvPw, IDX_QKV_BN2_W, IDX_QKV_BN2_B, IDX_QKV_BN2_M, IDX_QKV_BN2_V, 512);
  bncomp(gQkvPw, hQkvPw, IDX_QKVBN_W, IDX_QKVBN_B, IDX_QKVBN_M, IDX_QKVBN_V, 512);
  bnaff(gProj1, hProj1, IDX_PROJ_BN1_W, IDX_PROJ_BN1_B, IDX_PROJ_BN1_M, IDX_PROJ_BN1_V, CC);
  bnaff(gProjPw, hProjPw, IDX_PROJ_BN2_W, IDX_PROJ_BN2_B, IDX_PROJ_BN2_M, IDX_PROJ_BN2_V, CC);
  bncomp(gProjPw, hProjPw, IDX_PROJBN_W, IDX_PROJBN_B, IDX_PROJBN_M, IDX_PROJBN_V, CC);
  bnaff(gConv, hConv, IDX_CONV_BN_W, IDX_CONV_BN_B, IDX_CONV_BN_M, IDX_CONV_BN_V, CC);
  bnaff(gFc1, hFc1, IDX_MLP_BN1_W, IDX_MLP_BN1_B, IDX_MLP_BN1_M, IDX_MLP_BN1_V, HIDD);
  k_bias_fold<<<(HIDD + 255) / 256, 256, 0, stream>>>(hFc1, gFc1, F(IDX_MLP_FC1B), HIDD);
  bnaff(gFc2, hFc2, IDX_MLP_BN2_W, IDX_MLP_BN2_B, IDX_MLP_BN2_M, IDX_MLP_BN2_V, CC);
  k_bias_fold<<<(CC + 255) / 256, 256, 0, stream>>>(hFc2, gFc2, F(IDX_MLP_FC2B), CC);

  auto gemm = [&](const _Float16* A, const _Float16* Bsp, float* Y,
                  const float* g, const float* h, int M, int K) {
    dim3 grid(NN / 128, M / 128, NIMG);
    k_gemm_wmma<<<grid, 256, 0, stream>>>(A, Bsp, Y, g, h, M, K);
  };

  const int nBNC = (int)(((size_t)BBB * NN * CC) / 256);       // 16384
  const int nBNH = (int)(((size_t)BBB * NN * HIDD) / 256);     // 65536

  // -- attention + conv branch --
  k_lif_x<<<nBNC, 256, 0, stream>>>(F(IDX_X), sHard, sSoft);
  gemm(wQkvC1, sHard, yb1, gQkv1, hQkv1, CC, CC);              // qkv.c1 + bn1
  k_dw_pad<<<dim3(NN * CC / 256, NIMG), 256, 0, stream>>>(yb1, F(IDX_QKV_DW), hQkv1, dw1);
  gemm(wQkvPw, dw1, yqkv, gQkvPw, hQkvPw, 512, CC);            // qkv.pw + bn2∘qkv_bn
  k_qk_gate<<<BBB * CC, 256, 0, stream>>>(yqkv, gate);
  k_vmul<<<nBNC, 256, 0, stream>>>(yqkv, gate, satt);
  gemm(wProjC1, satt, yb2, gProj1, hProj1, CC, CC);            // proj.c1 + bn1
  k_dw_pad<<<dim3(NN * CC / 256, NIMG), 256, 0, stream>>>(yb2, F(IDX_PROJ_DW), hProj1, dw2);
  gemm(wProjPw, dw2, apre, gProjPw, hProjPw, CC, CC);          // proj.pw + bn2∘proj_bn
  gemm(wConv, sSoft, cvout, gConv, hConv, CC, CC);             // conv branch + bn

  // -- residual 1 + mlp first LIF --
  k_residual<<<nBNC, 256, 0, stream>>>(F(IDX_X), apre, cvout, F(IDX_SCALE), xmid, s3);

  // -- mlp --
  gemm(wFc1, s3, ybn, gFc1, hFc1, HIDD, CC);                   // fc1 + bias + bn1
  k_lif_hid<<<nBNH, 256, 0, stream>>>(ybn, sf);
  k_mlp_dw_lifz<<<nBNH, 256, 0, stream>>>(sf, ybn, F(IDX_MLP_DWW), F(IDX_MLP_DWB), sz);
  gemm(wFc2, sz, mlpo, gFc2, hFc2, CC, HIDD);                  // fc2 + bias + bn2

  // -- residual 2 -> channel-first output --
  k_final<<<nBNC, 256, 0, stream>>>(xmid, mlpo, (float*)d_out);
}
